// MultiHeadDualAttention_48069273977531
// MI455X (gfx1250) — matrix operand order repaired and checked by reading, not compiled, not run on hardware
//
#include <hip/hip_runtime.h>

typedef __attribute__((ext_vector_type(16))) __bf16 bf16x16;
typedef __attribute__((ext_vector_type(8)))  __bf16 bf16x8;
typedef __attribute__((ext_vector_type(4)))  __bf16 bf16x4;
typedef __attribute__((ext_vector_type(8)))  float  f32x8;
typedef __attribute__((ext_vector_type(4)))  int    i32x4;

#define NTOK   4096
#define DMODEL 256
#define DATTN  512
#define NHEAD  8
#define DHEAD  64
#define SCALE  0.125f   // 64^-0.5

// LDS row strides (elements). Multiples of 8 bf16 (=16B) keep ds_load_b128
// alignment; 40 -> 20-dword bank stride, 72 -> 36-dword: gcd(.,64)=4, so the
// 16 lanes of a fragment read hit 16 distinct bank groups.
#define LDA_G  40   // GEMM A tile stride   (128 x 40)
#define LDB_G  40   // GEMM W^T tile stride (64 x 40)
#define LDF    72   // flash Q/K/P and V^T tile stride (64 x 72)
#define LDS_S  65   // flash fp32 score stride (odd -> conflict-free row scans)

// ---------------------------------------------------------------------------
// CDNA5 async global->LDS staging (ASYNCcnt path), with portable fallback.
// Builtin signature (from toolchain diagnostic): 
//   void __builtin_amdgcn_global_load_async_to_lds_b128(
//        int4 addrspace(1)* gsrc, int4 addrspace(3)* ldst, imm int, imm int)
// ---------------------------------------------------------------------------
#if defined(__has_builtin)
#  if __has_builtin(__builtin_amdgcn_global_load_async_to_lds_b128) && \
      __has_builtin(__builtin_amdgcn_s_wait_asynccnt)
#    define HAVE_ASYNC_LDS 1
#  endif
#endif
#ifndef HAVE_ASYNC_LDS
#  define HAVE_ASYNC_LDS 0
#endif

// copy one 16-byte chunk per lane: LDS[dst] = GLOBAL[src]
static __device__ inline void cp16_g2l(__bf16* dst_lds, const __bf16* src_g) {
#if HAVE_ASYNC_LDS
  __builtin_amdgcn_global_load_async_to_lds_b128(
      (__attribute__((address_space(1))) i32x4*)src_g,
      (__attribute__((address_space(3))) i32x4*)dst_lds, 0, 0);
#else
  *(bf16x8*)dst_lds = *(const bf16x8*)src_g;
#endif
}
static __device__ inline void cp16_wait() {
#if HAVE_ASYNC_LDS
  __builtin_amdgcn_s_wait_asynccnt(0);
#endif
}

// ---------------------------------------------------------------------------
// WMMA helpers (wave32, 16x16x32 bf16 -> f32)
// ---------------------------------------------------------------------------
static __device__ inline f32x8 wmma_bf16(bf16x16 a, bf16x16 b, f32x8 c) {
  return __builtin_amdgcn_wmma_f32_16x16x32_bf16(
      /*neg_a=*/false, a, /*neg_b=*/false, b,
      /*c_mod=*/(short)0, c, /*reuse_a=*/false, /*reuse_b=*/false);
}

// A fragment (16 x K=32, row-major MxK in LDS).  Contiguous-k A order (not ISA table order):
// lanes 0-15 hold M=lane; elem j -> K=half*16+j (matches frag_B_tr)
static __device__ inline bf16x16 frag_A_rm(const __bf16* p, int ld, int lane) {
  const int row = lane & 15, half = lane >> 4;
  const __bf16* r = p + row * ld + half * 16;
  const bf16x8 lo = *(const bf16x8*)(r);
  const bf16x8 hi = *(const bf16x8*)(r + 8);
  bf16x16 f;
#pragma unroll
  for (int j = 0; j < 8; ++j) { f[j] = lo[j]; f[j + 8] = hi[j]; }
  return f;
}

// B fragment (K=32 x 16) from *transposed* storage: memory holds B^T row-major
// (N x K, K contiguous): lane (col, half) reads 16 consecutive bf16 -> 2x b128.
// Contiguous-k B order (as frag_A_rm): elem j -> K = half*16 + j, column N striped across lanes.
static __device__ inline bf16x16 frag_B_tr(const __bf16* p, int ld, int lane) {
  const int col = lane & 15, half = lane >> 4;
  const __bf16* r = p + col * ld + half * 16;
  const bf16x8 lo = *(const bf16x8*)(r);
  const bf16x8 hi = *(const bf16x8*)(r + 8);
  bf16x16 f;
#pragma unroll
  for (int j = 0; j < 8; ++j) { f[j] = lo[j]; f[j + 8] = hi[j]; }
  return f;
}

// ---------------------------------------------------------------------------
// GEMM + bias:  C[M,N] = A[M,K] @ W[K,N] + bias   (A fp32 or bf16, C fp32/bf16)
// block = 256 threads (8 waves); block tile 128x64; wave -> 16x64 strip.
// W tile is staged TRANSPOSED into LDS so B fragments are contiguous.
// ---------------------------------------------------------------------------
static __device__ inline void stage4(__bf16* dst, const float* src) {
  const float4 v = *(const float4*)src;
  bf16x4 o; o[0] = (__bf16)v.x; o[1] = (__bf16)v.y; o[2] = (__bf16)v.z; o[3] = (__bf16)v.w;
  *(bf16x4*)dst = o;
}
static __device__ inline void stage4(__bf16* dst, const __bf16* src) {
  *(bf16x4*)dst = *(const bf16x4*)src;
}

template <typename TA, typename TC>
__global__ __launch_bounds__(256)
void gemm_bias_bf16wmma(const TA* __restrict__ A, const float* __restrict__ W,
                        const float* __restrict__ bias, TC* __restrict__ C,
                        int M, int N, int K) {
  __shared__ __bf16 As[128 * LDA_G];  // 10 KB
  __shared__ __bf16 Wt[64 * LDB_G];   // 5 KB   (W^T: [n][k])
  const int tid  = threadIdx.x;
  const int wave = tid >> 5, lane = tid & 31;
  const int row0 = blockIdx.y * 128, col0 = blockIdx.x * 64;

  f32x8 acc[4] = {{}, {}, {}, {}};

  for (int k0 = 0; k0 < K; k0 += 32) {
#pragma unroll
    for (int i = 0; i < 4; ++i) {               // 128x32 A tile, row-major
      const int e = i * 1024 + tid * 4;
      const int r = e >> 5, c = e & 31;
      stage4(&As[r * LDA_G + c], &A[(size_t)(row0 + r) * K + k0 + c]);
    }
#pragma unroll
    for (int i = 0; i < 2; ++i) {               // 32x64 W tile -> transposed
      const int e = i * 1024 + tid * 4;
      const int r = e >> 6, c = e & 63;         // r = k index, c = n index
      const float4 v = *(const float4*)&W[(size_t)(k0 + r) * N + col0 + c];
      Wt[(c + 0) * LDB_G + r] = (__bf16)v.x;
      Wt[(c + 1) * LDB_G + r] = (__bf16)v.y;
      Wt[(c + 2) * LDB_G + r] = (__bf16)v.z;
      Wt[(c + 3) * LDB_G + r] = (__bf16)v.w;
    }
    __syncthreads();

    const bf16x16 a = frag_A_rm(&As[wave * 16 * LDA_G], LDA_G, lane);
#pragma unroll
    for (int t = 0; t < 4; ++t)
      acc[t] = wmma_bf16(a, frag_B_tr(&Wt[t * 16 * LDB_G], LDB_G, lane), acc[t]);
    __syncthreads();
  }

  const int coln = lane & 15, half = lane >> 4;
#pragma unroll
  for (int t = 0; t < 4; ++t) {
    const int col = col0 + t * 16 + coln;
    const float bv = bias[col];
#pragma unroll
    for (int r = 0; r < 8; ++r) {
      const int row = row0 + wave * 16 + half * 8 + r;   // C-frag: M = r + 8*half
      C[(size_t)row * N + col] = (TC)(acc[t][r] + bv);
    }
  }
}

// ---------------------------------------------------------------------------
// Flash attention, bf16 QKV (laid out [NTOK, DATTN], head = column slab h*64),
// fp32 online softmax, bf16 output.  block = 128 threads (4 waves), each block
// owns a 64-row query tile; wave w owns rows [w*16, w*16+16).
// Q/K tiles staged via CDNA5 async global->LDS; V staged transposed manually.
// ---------------------------------------------------------------------------
__global__ __launch_bounds__(128)
void flash_attn_bf16(const __bf16* __restrict__ Q, const __bf16* __restrict__ K,
                     const __bf16* __restrict__ V, __bf16* __restrict__ O) {
  __shared__ __bf16 Qs[64 * LDF];              // 9 KB
  __shared__ __bf16 Ks[64 * LDF];              // 9 KB  ([n][d])
  __shared__ __bf16 Vt[64 * LDF];              // 9 KB  (V^T: [d][n])
  __shared__ __bf16 Ps[64 * LDF];              // 9 KB
  __shared__ float  Ss[64 * LDS_S];            // 16.25 KB (scaled logits)
  __shared__ float  mrow[64], lrow[64], arow[64], mcur[64];
  __shared__ float  pmax[128], psum[128];

  const int tid  = threadIdx.x;
  const int wave = tid >> 5, lane = tid & 31;
  const int coln = lane & 15, half = lane >> 4;
  const int h    = blockIdx.y;
  const int q0   = blockIdx.x * 64;

  const __bf16* Qg = Q + h * DHEAD;
  const __bf16* Kg = K + h * DHEAD;
  const __bf16* Vg = V + h * DHEAD;
  __bf16*       Og = O + h * DHEAD;

#pragma unroll
  for (int i = 0; i < 4; ++i) {                // stage Q tile (64x64) async
    const int idx = i * 128 + tid;             // 16-byte chunk id (512 total)
    const int r = idx >> 3, c8 = (idx & 7) * 8;
    cp16_g2l(&Qs[r * LDF + c8], &Qg[(size_t)(q0 + r) * DATTN + c8]);
  }
  if (tid < 64) { mrow[tid] = -1e30f; lrow[tid] = 0.f; }

  f32x8 o[4] = {{}, {}, {}, {}};
  cp16_wait();
  __syncthreads();

  for (int j0 = 0; j0 < NTOK; j0 += 64) {
#pragma unroll
    for (int i = 0; i < 4; ++i) {              // stage K tile (64x64) async
      const int idx = i * 128 + tid;
      const int r = idx >> 3, c8 = (idx & 7) * 8;
      cp16_g2l(&Ks[r * LDF + c8], &Kg[(size_t)(j0 + r) * DATTN + c8]);
    }
#pragma unroll
    for (int i = 0; i < 8; ++i) {              // stage V tile -> transposed
      const int e = i * 512 + tid * 4;
      const int r = e >> 6, c = e & 63;
      const bf16x4 vv = *(const bf16x4*)&Vg[(size_t)(j0 + r) * DATTN + c];
      Vt[(c + 0) * LDF + r] = vv[0];
      Vt[(c + 1) * LDF + r] = vv[1];
      Vt[(c + 2) * LDF + r] = vv[2];
      Vt[(c + 3) * LDF + r] = vv[3];
    }
    if (j0 + 64 < NTOK)                        // pull next K tile toward L2
      __builtin_prefetch(&Kg[(size_t)(j0 + 64 + (tid >> 1)) * DATTN + (tid & 1) * 32], 0, 1);
    cp16_wait();
    __syncthreads();

    // ---- S = (Q K^T) * SCALE -------------------------------------------
#pragma unroll
    for (int t = 0; t < 4; ++t) {
      f32x8 s = {};
#pragma unroll
      for (int kk = 0; kk < 2; ++kk) {
        const bf16x16 a = frag_A_rm(&Qs[wave * 16 * LDF + kk * 32], LDF, lane);
        const bf16x16 b = frag_B_tr(&Ks[t * 16 * LDF + kk * 32], LDF, lane);
        s = wmma_bf16(a, b, s);
      }
#pragma unroll
      for (int r = 0; r < 8; ++r)
        Ss[(wave * 16 + half * 8 + r) * LDS_S + t * 16 + coln] = s[r] * SCALE;
    }
    __syncthreads();

    // ---- online softmax: 2 threads per row -----------------------------
    {
      const int row = tid >> 1, hh = tid & 1;
      const float* sr = &Ss[row * LDS_S + hh * 32];
      float mx = -1e30f;
#pragma unroll 8
      for (int x = 0; x < 32; ++x) mx = fmaxf(mx, sr[x]);
      pmax[tid] = mx;
    }
    __syncthreads();
    if (tid < 64) {
      const float mold = mrow[tid];
      const float mx = fmaxf(mold, fmaxf(pmax[2 * tid], pmax[2 * tid + 1]));
      mrow[tid] = mx;
      mcur[tid] = mx;
      arow[tid] = __expf(mold - mx);
    }
    __syncthreads();
    {
      const int row = tid >> 1, hh = tid & 1;
      const float mx = mcur[row];
      const float* sr = &Ss[row * LDS_S + hh * 32];
      __bf16* pr = &Ps[row * LDF + hh * 32];
      float ps = 0.f;
#pragma unroll
      for (int x4 = 0; x4 < 8; ++x4) {         // vectorized P writes (b64)
        bf16x4 pk;
#pragma unroll
        for (int u = 0; u < 4; ++u) {
          const float p = __expf(sr[x4 * 4 + u] - mx);
          pk[u] = (__bf16)p;
          ps += p;
        }
        *(bf16x4*)&pr[x4 * 4] = pk;
      }
      psum[tid] = ps;
    }
    __syncthreads();
    if (tid < 64)
      lrow[tid] = lrow[tid] * arow[tid] + psum[2 * tid] + psum[2 * tid + 1];

    // ---- O = O * alpha + P @ V -----------------------------------------
#pragma unroll
    for (int t = 0; t < 4; ++t)
#pragma unroll
      for (int r = 0; r < 8; ++r)
        o[t][r] *= arow[wave * 16 + half * 8 + r];
#pragma unroll
    for (int kk = 0; kk < 2; ++kk) {
      const bf16x16 a = frag_A_rm(&Ps[wave * 16 * LDF + kk * 32], LDF, lane);
#pragma unroll
      for (int t = 0; t < 4; ++t)
        o[t] = wmma_bf16(a, frag_B_tr(&Vt[t * 16 * LDF + kk * 32], LDF, lane), o[t]);
    }
    __syncthreads();
  }

  // ---- epilogue: O /= l, store bf16 ------------------------------------
#pragma unroll
  for (int t = 0; t < 4; ++t)
#pragma unroll
    for (int r = 0; r < 8; ++r) {
      const int rl = wave * 16 + half * 8 + r;
      Og[(size_t)(q0 + rl) * DATTN + t * 16 + coln] = (__bf16)(o[t][r] / lrow[rl]);
    }
}

// ---------------------------------------------------------------------------
// Host launch
// ---------------------------------------------------------------------------
extern "C" void kernel_launch(void* const* d_in, const int* in_sizes, int n_in,
                              void* d_out, int out_size, void* d_ws, size_t ws_size,
                              hipStream_t stream) {
  (void)in_sizes; (void)n_in; (void)out_size; (void)ws_size;
  const float* k1    = (const float*)d_in[0];
  const float* v1    = (const float*)d_in[1];
  const float* k2    = (const float*)d_in[2];
  const float* v2    = (const float*)d_in[3];
  const float* wk1_w = (const float*)d_in[4];
  const float* wk1_b = (const float*)d_in[5];
  const float* wv1_w = (const float*)d_in[6];
  const float* wv1_b = (const float*)d_in[7];
  const float* wk2_w = (const float*)d_in[8];
  const float* wk2_b = (const float*)d_in[9];
  const float* wv2_w = (const float*)d_in[10];
  const float* wv2_b = (const float*)d_in[11];
  const float* wo1_w = (const float*)d_in[12];
  const float* wo1_b = (const float*)d_in[13];
  const float* wo2_w = (const float*)d_in[14];
  const float* wo2_b = (const float*)d_in[15];

  const size_t PBYTES = (size_t)NTOK * DATTN * 2;   // one bf16 [4096,512] = 4 MB
  char* ws = (char*)d_ws;
  __bf16* k1p = (__bf16*)(ws + 0 * PBYTES);
  __bf16* v1p = (__bf16*)(ws + 1 * PBYTES);
  __bf16* k2p = (__bf16*)(ws + 2 * PBYTES);
  __bf16* v2p = (__bf16*)(ws + 3 * PBYTES);
  __bf16* o1  = (__bf16*)(ws + 4 * PBYTES);
  __bf16* o2  = (__bf16*)(ws + 5 * PBYTES);

  // 1) input projections (fp32 in -> bf16 out)
  const dim3 blk(256);
  const dim3 gproj(DATTN / 64, NTOK / 128);
  gemm_bias_bf16wmma<float, __bf16><<<gproj, blk, 0, stream>>>(k1, wk1_w, wk1_b, k1p, NTOK, DATTN, DMODEL);
  gemm_bias_bf16wmma<float, __bf16><<<gproj, blk, 0, stream>>>(v1, wv1_w, wv1_b, v1p, NTOK, DATTN, DMODEL);
  gemm_bias_bf16wmma<float, __bf16><<<gproj, blk, 0, stream>>>(k2, wk2_w, wk2_b, k2p, NTOK, DATTN, DMODEL);
  gemm_bias_bf16wmma<float, __bf16><<<gproj, blk, 0, stream>>>(v2, wv2_w, wv2_b, v2p, NTOK, DATTN, DMODEL);

  // 2) dual-direction attention as two flash passes (no N x N materialization)
  //    o2[n] = softmax_m(attn[n,:]) @ v2p      -> Q=k1p, K=k2p, V=v2p
  //    o1[m] = softmax_n(attn[:,m]) @ v1p      -> Q=k2p, K=k1p, V=v1p
  const dim3 gfl(NTOK / 64, NHEAD);
  flash_attn_bf16<<<gfl, dim3(128), 0, stream>>>(k1p, k2p, v2p, o2);
  flash_attn_bf16<<<gfl, dim3(128), 0, stream>>>(k2p, k1p, v1p, o1);

  // 3) output projections (bf16 in -> fp32 out, concatenated o1 then o2)
  float* out1 = (float*)d_out;
  float* out2 = out1 + (size_t)NTOK * DMODEL;
  const dim3 gout(DMODEL / 64, NTOK / 128);
  gemm_bias_bf16wmma<__bf16, float><<<gout, blk, 0, stream>>>(o1, wo1_w, wo1_b, out1, NTOK, DMODEL, DATTN);
  gemm_bias_bf16wmma<__bf16, float><<<gout, blk, 0, stream>>>(o2, wo2_w, wo2_b, out2, NTOK, DMODEL, DATTN);
}
